// HetSAGE_19567871000709
// MI455X (gfx1250) — matrix operand order repaired and checked by reading, not compiled, not run on hardware
//
#include <hip/hip_runtime.h>

#define DIM 256
#define KTOT 512

typedef __attribute__((ext_vector_type(16))) __bf16 v16bf;
typedef __attribute__((ext_vector_type(8)))  float  v8f;

// f32 -> bf16 round-to-nearest-even
__device__ __forceinline__ __bf16 f2bf(float f) {
  unsigned u = __builtin_bit_cast(unsigned, f);
  unsigned r = (u + 0x7FFFu + ((u >> 16) & 1u)) >> 16;
  unsigned short s = (unsigned short)r;
  return __builtin_bit_cast(__bf16, s);
}

// A fragment (16x32 bf16, contiguous K order): lane holds row = lane%16, half = lane/16.
// elements 0..7  <-> K = half*16 + i     (p0 = row + k0 + half*16)
// elements 8..15 <-> K = half*16 + 8 + i (p1 = row + k0 + half*16 + 8)
__device__ __forceinline__ v16bf make_a_frag(const float* __restrict__ p0,
                                             const float* __restrict__ p1,
                                             float sc) {
  float4 x0 = reinterpret_cast<const float4*>(p0)[0];
  float4 x1 = reinterpret_cast<const float4*>(p0)[1];
  float4 y0 = reinterpret_cast<const float4*>(p1)[0];
  float4 y1 = reinterpret_cast<const float4*>(p1)[1];
  v16bf r;
  r[0]  = f2bf(x0.x * sc); r[1]  = f2bf(x0.y * sc);
  r[2]  = f2bf(x0.z * sc); r[3]  = f2bf(x0.w * sc);
  r[4]  = f2bf(x1.x * sc); r[5]  = f2bf(x1.y * sc);
  r[6]  = f2bf(x1.z * sc); r[7]  = f2bf(x1.w * sc);
  r[8]  = f2bf(y0.x * sc); r[9]  = f2bf(y0.y * sc);
  r[10] = f2bf(y0.z * sc); r[11] = f2bf(y0.w * sc);
  r[12] = f2bf(y1.x * sc); r[13] = f2bf(y1.y * sc);
  r[14] = f2bf(y1.z * sc); r[15] = f2bf(y1.w * sc);
  return r;
}

// B fragment (32x16 bf16, contiguous K order): lane holds col = lane%16,
// elements i <-> K = half*16 + i : one contiguous 32B run in K-major storage.
__device__ __forceinline__ v16bf load_b_frag(const __bf16* __restrict__ p) {
  return *reinterpret_cast<const v16bf*>(p);
}

__device__ __forceinline__ void zero_acc(v8f* acc) {
#pragma unroll
  for (int t = 0; t < 16; ++t)
#pragma unroll
    for (int i = 0; i < 8; ++i) acc[t][i] = 0.0f;
}

// ---------------- prep kernels ----------------

__global__ void __launch_bounds__(256) k_count(const int* __restrict__ dst,
                                               float* __restrict__ cnt, int E_) {
  int e = blockIdx.x * 256 + threadIdx.x;
  if (e < E_) atomicAdd(cnt + dst[e], 1.0f);
}

__global__ void __launch_bounds__(256) k_inv(const float* __restrict__ cnt,
                                             float* __restrict__ inv, int n) {
  int i = blockIdx.x * 256 + threadIdx.x;
  if (i < n) inv[i] = 1.0f / fmaxf(cnt[i], 1.0f);
}

// W [K,C] f32 row-major -> Wt [C,K] bf16 (K-major per column)
__global__ void __launch_bounds__(256) k_wtrans(const float* __restrict__ W,
                                                __bf16* __restrict__ Wt,
                                                int K, int C) {
  int t = blockIdx.x * 256 + threadIdx.x;
  if (t >= K * C) return;
  int k = t / C, c = t % C;
  Wt[(size_t)c * K + k] = f2bf(W[t]);
}

// agg[dst] += h[src] (sum; mean applied later via inv)
__global__ void __launch_bounds__(256) k_scatter(const float* __restrict__ h,
                                                 const int* __restrict__ src,
                                                 const int* __restrict__ dst,
                                                 float* __restrict__ aggsum, int E_) {
  int e = blockIdx.x * 4 + (threadIdx.x >> 6);
  if (e >= E_) return;
  int c = (threadIdx.x & 63) * 4;
  int s = src[e], d = dst[e];
  float4 v = *reinterpret_cast<const float4*>(h + (size_t)s * DIM + c);
  float* out = aggsum + (size_t)d * DIM + c;
  atomicAdd(out + 0, v.x);
  atomicAdd(out + 1, v.y);
  atomicAdd(out + 2, v.z);
  atomicAdd(out + 3, v.w);
}

// ---------------- edge MLP: (et[E,32] @ Wem[32,256] + bem) scattered to addsum[dst] ----------------

__global__ void __launch_bounds__(256)
k_edge_mlp(const float* __restrict__ et, const __bf16* __restrict__ Wemt,
           const float* __restrict__ bem, const int* __restrict__ dst,
           float* __restrict__ addsum, int E_) {
  const int lane = threadIdx.x & 31;
  const int wv   = threadIdx.x >> 5;
  const int hf   = lane >> 4;
  const int lc   = lane & 15;
  const int eBase = blockIdx.x * 128 + wv * 16;

  int eA = eBase + lc; if (eA > E_ - 1) eA = E_ - 1;
  const float* base = et + (size_t)eA * 32;
  v16bf afrag = make_a_frag(base + hf * 16, base + hf * 16 + 8, 1.0f);

  v8f acc[16];
  zero_acc(acc);
#pragma unroll
  for (int t = 0; t < 16; ++t) {
    const __bf16* wb = Wemt + (size_t)(t * 16 + lc) * 32 + hf * 16;
    v16bf bfrag = load_b_frag(wb);
    acc[t] = __builtin_amdgcn_wmma_f32_16x16x32_bf16(
        false, afrag, false, bfrag, (short)0, acc[t], false, false);
  }

  // preload per-lane bias (loop-invariant across v) so atomics issue back-to-back
  float bem_r[16];
#pragma unroll
  for (int t = 0; t < 16; ++t) bem_r[t] = bem[t * 16 + lc];

#pragma unroll
  for (int v = 0; v < 8; ++v) {
    int e = eBase + v + 8 * hf;
    if (e >= E_) continue;
    int d = dst[e];
    float* out = addsum + (size_t)d * DIM;
#pragma unroll
    for (int t = 0; t < 16; ++t) {
      atomicAdd(out + t * 16 + lc, acc[t][v] + bem_r[t]);
    }
  }
}

// ---------------- fused concat-GEMM + bias + edge-bias + LayerNorm ----------------
// Y = LN( [Xself | aggsum*inv] @ W + bias + addsum*inv ) per row; wave = 16 rows x 256 cols.

__global__ void __launch_bounds__(256)
k_gemm_ln(const float* __restrict__ Xself, const float* __restrict__ aggsum,
          const float* __restrict__ inv, const __bf16* __restrict__ Wt,
          const float* __restrict__ bias, const float* __restrict__ addsum,
          const float* __restrict__ gamma, const float* __restrict__ beta,
          float* __restrict__ Y, int M) {
  const int lane = threadIdx.x & 31;
  const int wv   = threadIdx.x >> 5;
  const int hf   = lane >> 4;
  const int lc   = lane & 15;
  const int rowBase = blockIdx.x * 128 + wv * 16;

  int rA = rowBase + lc; if (rA > M - 1) rA = M - 1;
  const float aggsc = inv[rA];

  v8f acc[16];
  zero_acc(acc);

  for (int ks = 0; ks < 16; ++ks) {
    const int kk = ks * 32;
    const bool selfPart = (kk < 256);
    const float* srcp = selfPart ? Xself : aggsum;
    const int kof = selfPart ? kk : (kk - 256);
    const float sc = selfPart ? 1.0f : aggsc;
    const float* base = srcp + (size_t)rA * DIM + kof;
    v16bf afrag = make_a_frag(base + hf * 16, base + hf * 16 + 8, sc);
#pragma unroll
    for (int t = 0; t < 16; ++t) {
      const __bf16* wb = Wt + (size_t)(t * 16 + lc) * KTOT + kk + hf * 16;
      v16bf bfrag = load_b_frag(wb);
      acc[t] = __builtin_amdgcn_wmma_f32_16x16x32_bf16(
          false, afrag, false, bfrag, (short)0, acc[t], false, false);
    }
  }

  // preload per-lane output bias (loop-invariant across v)
  float bias_r[16];
#pragma unroll
  for (int t = 0; t < 16; ++t) bias_r[t] = bias[t * 16 + lc];

  // epilogue: bias + addsum*inv, LayerNorm across 256 cols (16 tiles x 16 lanes/half)
#pragma unroll
  for (int v = 0; v < 8; ++v) {
    int r = rowBase + v + 8 * hf;
    int rc = (r > M - 1) ? (M - 1) : r;
    float isc = inv[rc];
    float vals[16];
    float s = 0.0f, q = 0.0f;
#pragma unroll
    for (int t = 0; t < 16; ++t) {
      int col = t * 16 + lc;
      float y = acc[t][v] + bias_r[t] + addsum[(size_t)rc * DIM + col] * isc;
      vals[t] = y; s += y; q += y * y;
    }
    s += __shfl_xor(s, 1, 32);  q += __shfl_xor(q, 1, 32);
    s += __shfl_xor(s, 2, 32);  q += __shfl_xor(q, 2, 32);
    s += __shfl_xor(s, 4, 32);  q += __shfl_xor(q, 4, 32);
    s += __shfl_xor(s, 8, 32);  q += __shfl_xor(q, 8, 32);
    float mean = s * (1.0f / 256.0f);
    float var  = q * (1.0f / 256.0f) - mean * mean;
    float rstd = rsqrtf(var + 1e-5f);
    if (r < M) {
#pragma unroll
      for (int t = 0; t < 16; ++t) {
        int col = t * 16 + lc;
        Y[(size_t)r * DIM + col] = (vals[t] - mean) * rstd * gamma[col] + beta[col];
      }
    }
  }
}

// ---------------- host orchestration ----------------

extern "C" void kernel_launch(void* const* d_in, const int* in_sizes, int n_in,
                              void* d_out, int out_size, void* d_ws, size_t ws_size,
                              hipStream_t stream) {
  (void)n_in; (void)out_size; (void)ws_size;
  const float* h_user = (const float*)d_in[0];
  const float* h_item = (const float*)d_in[1];
  const float* et_ub  = (const float*)d_in[2];
  const float* et_bu  = (const float*)d_in[3];
  const float* W0_ub  = (const float*)d_in[4];
  const float* b0_ub  = (const float*)d_in[5];
  const float* W0_bu  = (const float*)d_in[6];
  const float* b0_bu  = (const float*)d_in[7];
  const float* W1_ub  = (const float*)d_in[8];
  const float* b1_ub  = (const float*)d_in[9];
  const float* W1_bu  = (const float*)d_in[10];
  const float* b1_bu  = (const float*)d_in[11];
  const float* Wem_ub = (const float*)d_in[12];
  const float* bem_ub = (const float*)d_in[13];
  const float* Wem_bu = (const float*)d_in[14];
  const float* bem_bu = (const float*)d_in[15];
  const float* g_u    = (const float*)d_in[16];
  const float* beta_u = (const float*)d_in[17];
  const float* g_i    = (const float*)d_in[18];
  const float* beta_i = (const float*)d_in[19];
  const int*   ei_ub  = (const int*)d_in[20];
  const int*   ei_bu  = (const int*)d_in[21];

  const int N = in_sizes[0] / DIM;
  const int E = in_sizes[2] / 32;
  const size_t ND = (size_t)N * DIM;

  char* w = (char*)d_ws;
  float* cntA = (float*)w; w += (size_t)N * 4;
  float* cntB = (float*)w; w += (size_t)N * 4;
  float* invA = (float*)w; w += (size_t)N * 4;
  float* invB = (float*)w; w += (size_t)N * 4;
  float* addI = (float*)w; w += ND * 4;   // edge-bias sums onto item dst (ub)
  float* addU = (float*)w; w += ND * 4;   // edge-bias sums onto user dst (bu)
  float* agg  = (float*)w; w += ND * 4;   // reusable feature-sum buffer
  float* hu1  = (float*)w; w += ND * 4;
  float* hi1  = (float*)w; w += ND * 4;
  __bf16* WtA0  = (__bf16*)w; w += (size_t)KTOT * DIM * 2;
  __bf16* WtB0  = (__bf16*)w; w += (size_t)KTOT * DIM * 2;
  __bf16* WtA1  = (__bf16*)w; w += (size_t)KTOT * DIM * 2;
  __bf16* WtB1  = (__bf16*)w; w += (size_t)KTOT * DIM * 2;
  __bf16* WemtA = (__bf16*)w; w += (size_t)32 * DIM * 2;
  __bf16* WemtB = (__bf16*)w; w += (size_t)32 * DIM * 2;

  float* hu_out = (float*)d_out;
  float* hi_out = (float*)d_out + ND;

  dim3 blk(256);
  const int cgrid = (E + 255) / 256;
  const int egrid = (E + 127) / 128;
  const int sgrid = (E + 3) / 4;
  const int ggrid = (N + 127) / 128;

  // zero counters + edge-bias accumulators (cntA/cntB and addI/addU are contiguous)
  hipMemsetAsync(cntA, 0, (size_t)2 * N * 4, stream);
  hipMemsetAsync(addI, 0, 2 * ND * 4, stream);

  k_count<<<cgrid, blk, 0, stream>>>(ei_ub + E, cntA, E);
  k_count<<<cgrid, blk, 0, stream>>>(ei_bu + E, cntB, E);
  k_inv<<<(2 * N + 255) / 256, blk, 0, stream>>>(cntA, invA, 2 * N);

  k_wtrans<<<(KTOT * DIM + 255) / 256, blk, 0, stream>>>(W0_ub, WtA0, KTOT, DIM);
  k_wtrans<<<(KTOT * DIM + 255) / 256, blk, 0, stream>>>(W0_bu, WtB0, KTOT, DIM);
  k_wtrans<<<(KTOT * DIM + 255) / 256, blk, 0, stream>>>(W1_ub, WtA1, KTOT, DIM);
  k_wtrans<<<(KTOT * DIM + 255) / 256, blk, 0, stream>>>(W1_bu, WtB1, KTOT, DIM);
  k_wtrans<<<(32 * DIM + 255) / 256, blk, 0, stream>>>(Wem_ub, WemtA, 32, DIM);
  k_wtrans<<<(32 * DIM + 255) / 256, blk, 0, stream>>>(Wem_bu, WemtB, 32, DIM);

  // edge-time bias sums (shared by both layers)
  k_edge_mlp<<<egrid, blk, 0, stream>>>(et_ub, WemtA, bem_ub, ei_ub + E, addI, E);
  k_edge_mlp<<<egrid, blk, 0, stream>>>(et_bu, WemtB, bem_bu, ei_bu + E, addU, E);

  // ---- layer 1 ----
  hipMemsetAsync(agg, 0, ND * 4, stream);
  k_scatter<<<sgrid, blk, 0, stream>>>(h_user, ei_ub, ei_ub + E, agg, E);
  k_gemm_ln<<<ggrid, blk, 0, stream>>>(h_item, agg, invA, WtA0, b0_ub, addI,
                                       g_i, beta_i, hi1, N);
  hipMemsetAsync(agg, 0, ND * 4, stream);
  k_scatter<<<sgrid, blk, 0, stream>>>(h_item, ei_bu, ei_bu + E, agg, E);
  k_gemm_ln<<<ggrid, blk, 0, stream>>>(h_user, agg, invB, WtB0, b0_bu, addU,
                                       g_u, beta_u, hu1, N);

  // ---- layer 2 (writes final outputs) ----
  hipMemsetAsync(agg, 0, ND * 4, stream);
  k_scatter<<<sgrid, blk, 0, stream>>>(hu1, ei_ub, ei_ub + E, agg, E);
  k_gemm_ln<<<ggrid, blk, 0, stream>>>(hi1, agg, invA, WtA1, b1_ub, addI,
                                       g_i, beta_i, hi_out, N);
  hipMemsetAsync(agg, 0, ND * 4, stream);
  k_scatter<<<sgrid, blk, 0, stream>>>(hi1, ei_bu, ei_bu + E, agg, E);
  k_gemm_ln<<<ggrid, blk, 0, stream>>>(hu1, agg, invB, WtB1, b1_bu, addU,
                                       g_u, beta_u, hu_out, N);
}
